// CWSemiPUContrastiveLoss_15994458210610
// MI455X (gfx1250) — compile-verified, matrix operand order checked by reading, not run on hardware
//
#include <hip/hip_runtime.h>

#define B_N 8192
#define TAU_F 0.07f

// Tensor Data Mover path: probe-confirmed builtins on gfx1250 toolchains.
// Host pass and non-TDM toolchains take the synchronous fallback.
#if defined(__HIP_DEVICE_COMPILE__) && __has_builtin(__builtin_amdgcn_tensor_load_to_lds) && \
    __has_builtin(__builtin_amdgcn_s_wait_tensorcnt)
#define HAVE_TDM 1
#else
#define HAVE_TDM 0
#endif

typedef int v8i __attribute__((ext_vector_type(8)));

// ---------------------------------------------------------------------------
// Kernel 1: exact label counts via V_WMMA_I32_16X16X64_IU8 (unchanged; wmma=3
// confirmed in the round-1 histogram). B == all-ones bytes makes
// D[m][n] = C + sum_k A[m][k]; summing column 0 over m yields the exact total
// of all 1024 packed indicator bytes, independent of byte->(m,k) placement.
// ---------------------------------------------------------------------------
__global__ __launch_bounds__(32) void count_labels_wmma(const int* __restrict__ labels,
                                                        int* __restrict__ counts) {
  const int lane = threadIdx.x;
  v8i cp = {}, cr = {}, cu = {};
  v8i bones;
#pragma unroll
  for (int d = 0; d < 8; ++d) bones[d] = 0x01010101;

  for (int t = 0; t < 8; ++t) {
    const int base = t * 1024 + lane * 32;
    v8i ap, ar, au;
#pragma unroll
    for (int d = 0; d < 8; ++d) {
      unsigned p = 0, r = 0, u = 0;
#pragma unroll
      for (int b = 0; b < 4; ++b) {
        const int lab = labels[base + d * 4 + b];
        p |= (unsigned)(lab == 1)  << (8 * b);
        r |= (unsigned)(lab == -1) << (8 * b);
        u |= (unsigned)(lab == 0)  << (8 * b);
      }
      ap[d] = (int)p; ar[d] = (int)r; au[d] = (int)u;
    }
    cp = __builtin_amdgcn_wmma_i32_16x16x64_iu8(false, ap, false, bones, cp, false, false);
    cr = __builtin_amdgcn_wmma_i32_16x16x64_iu8(false, ar, false, bones, cr, false, false);
    cu = __builtin_amdgcn_wmma_i32_16x16x64_iu8(false, au, false, bones, cu, false, false);
  }

  int sp = 0, sr = 0, su = 0;
#pragma unroll
  for (int d = 0; d < 8; ++d) { sp += cp[d]; sr += cr[d]; su += cu[d]; }
  sp = __shfl(sp, 0, 32) + __shfl(sp, 16, 32);
  sr = __shfl(sr, 0, 32) + __shfl(sr, 16, 32);
  su = __shfl(su, 0, 32) + __shfl(su, 16, 32);
  if (lane == 0) { counts[0] = sp; counts[1] = sr; counts[2] = su; }
}

#if HAVE_TDM
typedef unsigned int u32x4 __attribute__((ext_vector_type(4)));
typedef int i32x4 __attribute__((ext_vector_type(4)));
typedef int i32x8 __attribute__((ext_vector_type(8)));
typedef __attribute__((address_space(3))) void lds_void;

// 1-D tensor DMA: nfloats contiguous f32 from global -> LDS byte offset.
// D# per CDNA5 ISA: group0 = {count=1, lds_addr,
// global_addr[56:0], type=2}; group1 = {wg_mask=0, data_size=4B,
// tensor_dim0 = tile_dim0 = nfloats, tensor_dim1 = 1, tile_dim1/2 unused,
// tensor_dim0_stride = nfloats}; groups 2/3 zero (<=2D tensor).
__device__ __forceinline__ void tdm_load_1d(const float* gsrc, unsigned lds_byte_off,
                                            int nfloats) {
  const unsigned long long ga = (unsigned long long)(size_t)gsrc;
  u32x4 g0;
  i32x8 g1;
  i32x4 g2 = {0, 0, 0, 0};
  i32x4 g3 = {0, 0, 0, 0};
  g0[0] = 1u;                                               // count=1, user D#
  g0[1] = lds_byte_off;                                     // lds_addr (bytes)
  g0[2] = (unsigned)(ga & 0xffffffffull);                   // global_addr lo32
  g0[3] = (unsigned)((ga >> 32) & 0x01ffffffull) | (2u << 30);  // addr hi | type=2
  g1[0] = (int)(2u << 16);                                  // data_size = 4B
  g1[1] = (int)(((unsigned)nfloats & 0xffffu) << 16);       // tensor_dim0 lo16
  g1[2] = (int)((((unsigned)nfloats >> 16) & 0xffffu) | (1u << 16)); // dim0 hi16 | tensor_dim1=1
  g1[3] = (int)(((unsigned)nfloats & 0xffffu) << 16);       // tile_dim0
  g1[4] = 0;                                                // tile_dim1/2 unused
  g1[5] = nfloats;                                          // tensor_dim0_stride lo32
  g1[6] = 0;
  g1[7] = 0;
#if __has_include(<hip/amd_detail/amd_gfx1250_TDM.h>)
  // amdgpu-toolchain (clang-23) lane: 6-arg builtin
  i32x8 g4 = {0, 0, 0, 0, 0, 0, 0, 0};
  __builtin_amdgcn_tensor_load_to_lds(g0, g1, g2, g3, g4, 0);
#else
  // ROCm 7.2 (clang-22) lane: 5-arg builtin
  __builtin_amdgcn_tensor_load_to_lds(g0, g1, g2, g3, 0);
#endif
}
#endif  // HAVE_TDM

// ---------------------------------------------------------------------------
// Kernel 2: one workgroup (8 waves) per row.
// TDM path: each wave DMAs its 4KB slice of the sim row AND the w row into
// LDS (tensor_load_to_lds, TENSORcnt). Wait tensorcnt<=1 -> sim ready for the
// max/logsumexp passes while the w DMA is still in flight; wait tensorcnt==0
// only before the masked-sum pass. Both matrices flow HBM->LDS exactly once
// with no VGPR staging.
// ---------------------------------------------------------------------------
__global__ __launch_bounds__(256) void row_loss_kernel(
    const float* __restrict__ sim, const float* __restrict__ alphas,
    const float* __restrict__ betas, const float* __restrict__ pi_a,
    const float* __restrict__ w, const int* __restrict__ labels,
    const int* __restrict__ counts, float* __restrict__ row_loss) {
  __shared__ float srow[B_N];
#if HAVE_TDM
  __shared__ float wldsrow[B_N];
#endif
  __shared__ float red[8];
  __shared__ float red4[8][4];
  __shared__ float bcast[2];

  const int i = blockIdx.x;
  const int tid = threadIdx.x;
  const int lane = tid & 31;
  const int wid = tid >> 5;
  const float invTau = 1.0f / TAU_F;

  const float* __restrict__ gsim = sim + (size_t)i * B_N;
  const float* __restrict__ gw = w + (size_t)i * B_N;

  float lmax = -3.4e38f;

#if HAVE_TDM
  {
    const unsigned sOff = (unsigned)(size_t)(lds_void*)(void*)srow;
    const unsigned wOff = (unsigned)(size_t)(lds_void*)(void*)wldsrow;
    const int slice = B_N / 8;  // 1024 floats per wave
    tdm_load_1d(gsim + wid * slice, sOff + (unsigned)(wid * slice) * 4u, slice);
    tdm_load_1d(gw + wid * slice, wOff + (unsigned)(wid * slice) * 4u, slice);
    __builtin_amdgcn_s_wait_tensorcnt(1);  // in-order: sim slice complete
  }
  __syncthreads();

  // ---- Pass A: row max (j != i) from LDS
  {
    const float4* __restrict__ sv = reinterpret_cast<const float4*>(srow);
    for (int v4 = tid; v4 < B_N / 4; v4 += 256) {
      const float4 x = sv[v4];
      const int j = v4 * 4;
      if (j + 0 != i) lmax = fmaxf(lmax, x.x);
      if (j + 1 != i) lmax = fmaxf(lmax, x.y);
      if (j + 2 != i) lmax = fmaxf(lmax, x.z);
      if (j + 3 != i) lmax = fmaxf(lmax, x.w);
    }
  }
#else
  // ---- Pass A (fallback): stage sim row to LDS with B128 loads, track max
  {
    const float4* __restrict__ simv = reinterpret_cast<const float4*>(gsim);
    float4* __restrict__ srowv = reinterpret_cast<float4*>(srow);
    for (int v4 = tid; v4 < B_N / 4; v4 += 256) {
      const float4 x = simv[v4];
      srowv[v4] = x;
      const int j = v4 * 4;
      if (j + 0 != i) lmax = fmaxf(lmax, x.x);
      if (j + 1 != i) lmax = fmaxf(lmax, x.y);
      if (j + 2 != i) lmax = fmaxf(lmax, x.z);
      if (j + 3 != i) lmax = fmaxf(lmax, x.w);
    }
  }
#endif

#pragma unroll
  for (int o = 16; o; o >>= 1) lmax = fmaxf(lmax, __shfl_xor(lmax, o, 32));
  if (lane == 0) red[wid] = lmax;
  __syncthreads();
  if (tid == 0) {
    float m = red[0];
#pragma unroll
    for (int k = 1; k < 8; ++k) m = fmaxf(m, red[k]);
    bcast[0] = m;
  }
  __syncthreads();
  const float ml = bcast[0] * invTau;  // max logit over j != i

  // ---- Pass B: Z = sum_{j!=i} exp(s_j/tau - ml)
#if !HAVE_TDM
  for (int j = tid * 16; j < B_N; j += 256 * 16)
    __builtin_prefetch(gw + j, 0, 1);  // warm the w stream (global_prefetch_b8)
#endif
  float z = 0.0f;
  for (int j = tid; j < B_N; j += 256) {
    const float e = __expf(srow[j] * invTau - ml);
    z += (j != i) ? e : 0.0f;
  }
#pragma unroll
  for (int o = 16; o; o >>= 1) z += __shfl_xor(z, o, 32);
  if (lane == 0) red[wid] = z;
  __syncthreads();
  if (tid == 0) {
    float t = 0.0f;
#pragma unroll
    for (int k = 0; k < 8; ++k) t += red[k];
    bcast[1] = ml + logf(t);  // logZ_i
  }
#if HAVE_TDM
  __builtin_amdgcn_s_wait_tensorcnt(0);  // w slice complete (overlapped with B)
#endif
  __syncthreads();
  const float logZ = bcast[1];

  // ---- Pass C: masked sums (branch-free)
#if HAVE_TDM
  const float* __restrict__ wsrc = wldsrow;
#else
  const float* __restrict__ wsrc = gw;
#endif
  float s1 = 0.0f, ep = 0.0f, srn = 0.0f, eu = 0.0f;
  for (int j = tid; j < B_N; j += 256) {
    const float c = srow[j] * invTau - logZ;  // centered (<= ~0, exp safe)
    const float e = __expf(c);
    const float wj = wsrc[j];
    const int lab = labels[j];
    const float off = (j != i) ? 1.0f : 0.0f;
    const float isP = (lab == 1)  ? off : 0.0f;
    const float isR = (lab == -1) ? off : 0.0f;
    const float isU = (lab == 0)  ? off : 0.0f;
    s1  += isP * (alphas[j] * c);
    ep  += isP * e;
    srn += isR * (betas[j] * (wj * e));
    eu  += isU * (wj * e);
  }
#pragma unroll
  for (int o = 16; o; o >>= 1) {
    s1  += __shfl_xor(s1, o, 32);
    ep  += __shfl_xor(ep, o, 32);
    srn += __shfl_xor(srn, o, 32);
    eu  += __shfl_xor(eu, o, 32);
  }
  if (lane == 0) { red4[wid][0] = s1; red4[wid][1] = ep; red4[wid][2] = srn; red4[wid][3] = eu; }
  __syncthreads();

  if (tid == 0) {
    float S1 = 0, EP = 0, SRN = 0, EU = 0;
#pragma unroll
    for (int k = 0; k < 8; ++k) {
      S1 += red4[k][0]; EP += red4[k][1]; SRN += red4[k][2]; EU += red4[k][3];
    }
    const int labi = labels[i];
    const int cpn = counts[0] - (labi == 1 ? 1 : 0);
    const int crn = counts[1] - (labi == -1 ? 1 : 0);
    const int cun = counts[2] - (labi == 0 ? 1 : 0);

    const float Lpos = (cpn > 0) ? (-S1 / (float)cpn) : 0.0f;
    const float Lrn  = (crn > 0) ? (SRN / (float)crn) : 0.0f;
    const float E_U  = EU / (float)(cun > 0 ? cun : 1);
    const float E_P  = EP / (float)(cpn > 0 ? cpn : 1);
    const float pi   = fminf(fmaxf(pi_a[i], 1e-4f), 0.5f);
    const float deb  = (E_U - pi * E_P) / (1.0f - pi + 1e-8f);
    const float Lu   = (cun > 0 && cpn > 0) ? fmaxf(deb, 0.0f) : 0.0f;

    row_loss[i] = Lpos + Lrn + Lu;
  }
}

// ---------------------------------------------------------------------------
// Kernel 3: deterministic mean over the 8192 per-row losses.
// ---------------------------------------------------------------------------
__global__ __launch_bounds__(256) void finalize_mean(const float* __restrict__ row_loss,
                                                     float* __restrict__ out) {
  __shared__ float red[8];
  const int tid = threadIdx.x;
  const int lane = tid & 31, wid = tid >> 5;
  float v = 0.0f;
  for (int j = tid; j < B_N; j += 256) v += row_loss[j];
#pragma unroll
  for (int o = 16; o; o >>= 1) v += __shfl_xor(v, o, 32);
  if (lane == 0) red[wid] = v;
  __syncthreads();
  if (tid == 0) {
    float t = 0.0f;
#pragma unroll
    for (int k = 0; k < 8; ++k) t += red[k];
    out[0] = t / (float)B_N;
  }
}

extern "C" void kernel_launch(void* const* d_in, const int* in_sizes, int n_in,
                              void* d_out, int out_size, void* d_ws, size_t ws_size,
                              hipStream_t stream) {
  const float* sim    = (const float*)d_in[0];
  const float* alphas = (const float*)d_in[1];
  const float* betas  = (const float*)d_in[2];
  const float* pi_a   = (const float*)d_in[3];
  const float* w      = (const float*)d_in[4];
  const int*   labels = (const int*)d_in[5];
  float* out = (float*)d_out;

  float* row_loss = (float*)d_ws;                              // 8192 floats
  int*   counts   = (int*)((char*)d_ws + B_N * sizeof(float)); // 3 ints

  count_labels_wmma<<<1, 32, 0, stream>>>(labels, counts);
  row_loss_kernel<<<B_N, 256, 0, stream>>>(sim, alphas, betas, pi_a, w, labels,
                                           counts, row_loss);
  finalize_mean<<<1, 256, 0, stream>>>(row_loss, out);
}
